// QModule_64415919506092
// MI455X (gfx1250) — 8-bit matrix operand order repaired and checked by reading, compiled, not run on hardware
//
#include <hip/hip_runtime.h>

typedef __attribute__((ext_vector_type(8))) int v8i;

#define M_TOT 32768
#define N_TOT 1024
#define K_TOT 1024
#define QMAXF 127.0f

__global__ __launch_bounds__(256) void quant_act_kernel(
    const float* __restrict__ x, const float* __restrict__ act_scale,
    unsigned* __restrict__ xq4) {
  const float inv_s = 1.0f / act_scale[0];
  const long i = (long)blockIdx.x * blockDim.x + threadIdx.x;
  const float4 v = ((const float4*)x)[i];
  const int a = (int)rintf(fminf(fmaxf(v.x * inv_s, -QMAXF), QMAXF));
  const int b = (int)rintf(fminf(fmaxf(v.y * inv_s, -QMAXF), QMAXF));
  const int c = (int)rintf(fminf(fmaxf(v.z * inv_s, -QMAXF), QMAXF));
  const int d = (int)rintf(fminf(fmaxf(v.w * inv_s, -QMAXF), QMAXF));
  xq4[i] = (unsigned)(a & 255) | ((unsigned)(b & 255) << 8) |
           ((unsigned)(c & 255) << 16) | ((unsigned)(d & 255) << 24);
}

__global__ __launch_bounds__(256) void quant_w_kernel(
    const float* __restrict__ w, const float* __restrict__ act_scale,
    unsigned* __restrict__ qw4, float* __restrict__ wscale) {
  __shared__ float red[256];
  const int row = blockIdx.x;
  const int t = threadIdx.x;
  const float4 v = ((const float4*)(w + (long)row * K_TOT))[t];
  red[t] = fmaxf(fmaxf(fabsf(v.x), fabsf(v.y)), fmaxf(fabsf(v.z), fabsf(v.w)));
  __syncthreads();
#pragma unroll
  for (int off = 128; off > 0; off >>= 1) {
    if (t < off) red[t] = fmaxf(red[t], red[t + off]);
    __syncthreads();
  }
  const float absmax = red[0];
  if (t == 0) wscale[row] = absmax * act_scale[0] / QMAXF;
  const float inv = QMAXF / absmax;
  const int a = (int)rintf(v.x * inv);
  const int b = (int)rintf(v.y * inv);
  const int c = (int)rintf(v.z * inv);
  const int d = (int)rintf(v.w * inv);
  qw4[(long)row * (K_TOT / 4) + t] =
      (unsigned)(a & 255) | ((unsigned)(b & 255) << 8) |
      ((unsigned)(c & 255) << 16) | ((unsigned)(d & 255) << 24);
}

union AFrag { unsigned long long u64[4]; v8i v; };
union BFrag { uint4 q[2]; v8i v; };

__global__ __launch_bounds__(256) void gemm_iu8_kernel(
    const signed char* __restrict__ xq, const signed char* __restrict__ qw,
    const float* __restrict__ wscale, const float* __restrict__ bias,
    float* __restrict__ y) {
  const int lane = threadIdx.x & 31;
  const int wid = threadIdx.x >> 5;
  const int wave_m = wid & 3;
  const int wave_n = wid >> 2;
  const int mbase = blockIdx.y * 128 + wave_m * 32;
  const int nbase = blockIdx.x * 128 + wave_n * 64;
  const int ml = lane & 15;
  const int hi = lane >> 4;
  const int koffA = hi * 16;
  const int koffB = hi * 16;

  v8i acc[2][4] = {};

  const signed char* aRow0 = xq + (long)(mbase + ml) * K_TOT + koffA;
  const signed char* aRow1 = aRow0 + 16 * K_TOT;
  const signed char* bCol[4];
#pragma unroll
  for (int tn = 0; tn < 4; ++tn)
    bCol[tn] = qw + (long)(nbase + tn * 16 + ml) * K_TOT + koffB;

  for (int k0 = 0; k0 < K_TOT; k0 += 64) {
    AFrag a[2];
    {
      const unsigned long long* p0 = (const unsigned long long*)(aRow0 + k0);
      const unsigned long long* p1 = (const unsigned long long*)(aRow1 + k0);
      a[0].u64[0] = p0[0]; a[0].u64[1] = p0[1];
      a[0].u64[2] = p0[4]; a[0].u64[3] = p0[5];
      a[1].u64[0] = p1[0]; a[1].u64[1] = p1[1];
      a[1].u64[2] = p1[4]; a[1].u64[3] = p1[5];
    }
    BFrag b[4];
#pragma unroll
    for (int tn = 0; tn < 4; ++tn) {
      const uint4* p = (const uint4*)(bCol[tn] + k0);
      b[tn].q[0] = p[0];
      b[tn].q[1] = p[2];
    }
#pragma unroll
    for (int tm = 0; tm < 2; ++tm)
#pragma unroll
      for (int tn = 0; tn < 4; ++tn)
        acc[tm][tn] = __builtin_amdgcn_wmma_i32_16x16x64_iu8(
             true, a[tm].v,  true, b[tn].v, acc[tm][tn],
             false,  false);
  }

  const int mo = hi * 8;
#pragma unroll
  for (int tn = 0; tn < 4; ++tn) {
    const int n = nbase + tn * 16 + ml;
    const float ws = wscale[n];
    const float bi = bias[n];
#pragma unroll
    for (int tm = 0; tm < 2; ++tm) {
      const int row0 = mbase + tm * 16 + mo;
#pragma unroll
      for (int r = 0; r < 8; ++r)
        y[(long)(row0 + r) * N_TOT + n] = (float)acc[tm][tn][r] * ws + bi;
    }
  }
}

extern "C" void kernel_launch(void* const* d_in, const int* in_sizes, int n_in,
                              void* d_out, int out_size, void* d_ws,
                              size_t ws_size, hipStream_t stream) {
  const float* x = (const float*)d_in[0];
  const float* w = (const float*)d_in[1];
  const float* bias = (const float*)d_in[2];
  const float* act_scale = (const float*)d_in[3];
  float* y = (float*)d_out;

  signed char* xq = (signed char*)d_ws;
  signed char* qw = xq + (size_t)M_TOT * K_TOT;
  float* wscale = (float*)(qw + (size_t)N_TOT * K_TOT);

  quant_act_kernel<<<(M_TOT * K_TOT / 4) / 256, 256, 0, stream>>>(
      x, act_scale, (unsigned*)xq);
  quant_w_kernel<<<N_TOT, 256, 0, stream>>>(w, act_scale, (unsigned*)qw,
                                            wscale);
  dim3 grid(N_TOT / 128, M_TOT / 128);
  gemm_iu8_kernel<<<grid, 256, 0, stream>>>(xq, qw, wscale, bias, y);
}
